// MambaLayer_23304492548706
// MI455X (gfx1250) — compile-verified, matrix operand order checked by reading, not run on hardware
//
#include <hip/hip_runtime.h>
#include <hip/hip_bf16.h>

#define B_  2
#define L_  1024
#define DM_ 2048
#define DI_ 4096
#define DN_ 16
#define DR_ 128
#define DC_ 4
#define ML_ (B_ * L_)        // 2048 rows in every GEMM
#define DX_ (DR_ + 2 * DN_)  // 160

typedef __attribute__((ext_vector_type(16))) __bf16        v16bf;
typedef __attribute__((ext_vector_type(8)))  float         v8f;
typedef __attribute__((ext_vector_type(4)))  unsigned int  v4u;

__device__ __forceinline__ unsigned short f32_to_bf16_rne(float x) {
    unsigned a = __float_as_uint(x);
    a = (a + 0x7FFFu + ((a >> 16) & 1u)) >> 16;
    return (unsigned short)a;
}
__device__ __forceinline__ float sigmoidf_(float x) {
    return 1.0f / (1.0f + __expf(-x));
}
__device__ __forceinline__ float softplusf_(float x) {
    return (x > 20.0f) ? x : log1pf(__expf(x));
}

union BfFrag { v16bf v; v4u q[2]; };

// ---------------------------------------------------------------------------
// C[M,N] = act( A[M,K] * W[N,K]^T + bias ), A/W bf16, C fp32.
// K-permutation trick: each lane loads 16 CONTIGUOUS bf16 at element offset
// k0 + 16*half for both A-row and W-row; the same (lane,vgpr,slot)->K mapping
// applies to both fragments so the dot product is exact.
// Wave tile: 32(M) x NT*16(N). NT is compile-time -> no guards in the K loop:
// per chunk, one load clause (2 A + NT B b128-pairs) then 2*NT WMMAs.
// grid: x = ceil(nstrips/8), y = M/32 ; block = 256 (8 waves)
// ---------------------------------------------------------------------------
template<int ACT, int NT>
__global__ __launch_bounds__(256) void wmma_gemm_bf16(
    const unsigned short* __restrict__ A, int lda,
    const unsigned short* __restrict__ W,           // N x K
    const float* __restrict__ bias,                 // ACT==1 only
    float* __restrict__ C, int ldc,
    int K, int nstrips, int ntile0)
{
    const int lane = threadIdx.x & 31;
    const int wave = threadIdx.x >> 5;
    const int half = lane >> 4;
    const int lr   = lane & 15;

    const int m32   = blockIdx.y;
    const int strip = blockIdx.x * 8 + wave;
    if (strip >= nstrips) return;                 // single uniform early-out
    const int tile0 = ntile0 + strip * NT;

    const int row0 = m32 * 32 + lr;
    const v4u* Aq0 = (const v4u*)(A + (size_t)row0 * lda);
    const v4u* Aq1 = (const v4u*)(A + (size_t)(row0 + 16) * lda);

    const v4u* Wq[NT];
#pragma unroll
    for (int t = 0; t < NT; ++t)
        Wq[t] = (const v4u*)(W + (size_t)((tile0 + t) * 16 + lr) * K);

    v8f acc[2][NT];
#pragma unroll
    for (int m = 0; m < 2; ++m)
#pragma unroll
        for (int t = 0; t < NT; ++t) acc[m][t] = (v8f){};

    for (int k0 = 0; k0 < K; k0 += 32) {
        const int qoff = (k0 + 16 * half) >> 3;   // v4u = 8 bf16
        BfFrag a0, a1, bfr[NT];
        a0.q[0] = Aq0[qoff]; a0.q[1] = Aq0[qoff + 1];
        a1.q[0] = Aq1[qoff]; a1.q[1] = Aq1[qoff + 1];
#pragma unroll
        for (int t = 0; t < NT; ++t) {
            bfr[t].q[0] = Wq[t][qoff];
            bfr[t].q[1] = Wq[t][qoff + 1];
        }
#pragma unroll
        for (int t = 0; t < NT; ++t) {
            acc[0][t] = __builtin_amdgcn_wmma_f32_16x16x32_bf16(
                false, a0.v, false, bfr[t].v, (short)0, acc[0][t], false, false);
            acc[1][t] = __builtin_amdgcn_wmma_f32_16x16x32_bf16(
                false, a1.v, false, bfr[t].v, (short)0, acc[1][t], false, false);
        }
    }

#pragma unroll
    for (int t = 0; t < NT; ++t) {
        const int col = (tile0 + t) * 16 + lr;
        const float bv = (ACT == 1) ? bias[col] : 0.0f;
#pragma unroll
        for (int m = 0; m < 2; ++m) {
#pragma unroll
            for (int v = 0; v < 8; ++v) {
                const int row = m32 * 32 + m * 16 + v + 8 * half;
                float val = acc[m][t][v];
                if (ACT == 1) val = softplusf_(val + bv);
                C[(size_t)row * ldc + col] = val;
            }
        }
    }
}

// host-side dispatcher: full 64-wide strips (NT=4) + one NT=2 remainder (N=160)
template<int ACT>
static void launch_gemm(const unsigned short* A, int lda,
                        const unsigned short* W, const float* bias,
                        float* C, int ldc, int N, int K, hipStream_t s)
{
    const unsigned m32_tiles = ML_ / 32;
    const int full = N / 64;
    if (full > 0) {
        dim3 grid((unsigned)((full + 7) / 8), m32_tiles);
        wmma_gemm_bf16<ACT, 4><<<grid, dim3(256), 0, s>>>(A, lda, W, bias, C, ldc,
                                                          K, full, 0);
    }
    const int rem = (N % 64) / 16;   // only the N=160 GEMM has rem (=2)
    if (rem == 2) {
        dim3 grid(1, m32_tiles);
        wmma_gemm_bf16<ACT, 2><<<grid, dim3(256), 0, s>>>(A, lda, W, bias, C, ldc,
                                                          K, 1, full * 4);
    }
}

// elementwise fp32 -> bf16 (RNE)
__global__ __launch_bounds__(256) void cvt_bf16_kernel(
    const float* __restrict__ s, unsigned short* __restrict__ d, size_t n)
{
    const size_t i = (size_t)blockIdx.x * blockDim.x + threadIdx.x;
    if (i < n) d[i] = f32_to_bf16_rne(s[i]);
}

// causal depthwise conv1d (k=4) on x = xz[..., :DI], then SiLU -> fp32 + bf16
__global__ __launch_bounds__(256) void conv_silu_kernel(
    const float* __restrict__ xz, const float* __restrict__ cw,
    float* __restrict__ xc, unsigned short* __restrict__ xcb)
{
    const size_t idx = (size_t)blockIdx.x * blockDim.x + threadIdx.x;
    if (idx >= (size_t)ML_ * DI_) return;
    const int d = (int)(idx % DI_);
    const int t = (int)(idx / DI_);
    const int l = t % L_;
    const int b = t / L_;
    float s = 0.0f;
#pragma unroll
    for (int w = 0; w < DC_; ++w) {
        const int ls = l + w - (DC_ - 1);
        if (ls >= 0)
            s += xz[((size_t)(b * L_ + ls)) * (2 * DI_) + d] * cw[d * DC_ + w];
    }
    s = s * sigmoidf_(s);
    xc[idx]  = s;
    xcb[idx] = f32_to_bf16_rne(s);
}

// selective scan: one thread per (b, d); h[DN] in registers; emits bf16 y
__global__ __launch_bounds__(256) void scan_kernel(
    const float* __restrict__ delta, const float* __restrict__ xdbl,
    const float* __restrict__ xc,    const float* __restrict__ xz,
    const float* __restrict__ A_log, const float* __restrict__ Dp,
    unsigned short* __restrict__ yb)
{
    const int gid = blockIdx.x * blockDim.x + threadIdx.x;
    if (gid >= B_ * DI_) return;
    const int d = gid % DI_;
    const int b = gid / DI_;

    float Arow[DN_];
#pragma unroll
    for (int n = 0; n < DN_; ++n) Arow[n] = -__expf(A_log[d * DN_ + n]);
    float h[DN_];
#pragma unroll
    for (int n = 0; n < DN_; ++n) h[n] = 0.0f;
    const float Dd = Dp[d];

    for (int l = 0; l < L_; ++l) {
        const size_t rt = (size_t)(b * L_ + l);
        const float dl = delta[rt * DI_ + d];
        const float u  = xc[rt * DI_ + d];
        const float* bc = xdbl + rt * DX_;
        const float dlu = dl * u;
        float yv = 0.0f;
#pragma unroll
        for (int n = 0; n < DN_; ++n) {
            const float dA = __expf(dl * Arow[n]);
            h[n] = dA * h[n] + dlu * bc[DR_ + n];
            yv  += h[n] * bc[DR_ + DN_ + n];
        }
        yv += u * Dd;
        const float zv = xz[rt * (2 * DI_) + DI_ + d];
        yv *= zv * sigmoidf_(zv);
        yb[rt * DI_ + d] = f32_to_bf16_rne(yv);
    }
}

static inline unsigned cdiv_(size_t a, unsigned b) { return (unsigned)((a + b - 1) / b); }

extern "C" void kernel_launch(void* const* d_in, const int* in_sizes, int n_in,
                              void* d_out, int out_size, void* d_ws, size_t ws_size,
                              hipStream_t stream) {
    const float* hs     = (const float*)d_in[0]; // (B,L,DM)
    const float* w_in   = (const float*)d_in[1]; // (2DI, DM)
    const float* w_conv = (const float*)d_in[2]; // (DI, DC)
    const float* w_xp   = (const float*)d_in[3]; // (DX, DI)
    const float* w_dt   = (const float*)d_in[4]; // (DI, DR)
    const float* b_dt   = (const float*)d_in[5]; // (DI)
    const float* A_log  = (const float*)d_in[6]; // (DI, DN)
    const float* Dp     = (const float*)d_in[7]; // (DI)
    const float* w_out  = (const float*)d_in[8]; // (DM, DI)
    float* out = (float*)d_out;

    // ---- workspace layout ----
    float* fws = (float*)d_ws;
    float* xz    = fws;                                  // ML x 2DI
    float* xc    = xz    + (size_t)ML_ * 2 * DI_;        // ML x DI
    float* xdbl  = xc    + (size_t)ML_ * DI_;            // ML x DX
    float* delta = xdbl  + (size_t)ML_ * DX_;            // ML x DI
    unsigned short* us = (unsigned short*)(delta + (size_t)ML_ * DI_);
    unsigned short* hs_bf    = us;                                  // ML x DM
    unsigned short* w_in_bf  = hs_bf   + (size_t)ML_ * DM_;         // 2DI x DM
    unsigned short* w_xp_bf  = w_in_bf + (size_t)2 * DI_ * DM_;     // DX x DI
    unsigned short* w_dt_bf  = w_xp_bf + (size_t)DX_ * DI_;         // DI x DR
    unsigned short* w_out_bf = w_dt_bf + (size_t)DI_ * DR_;         // DM x DI
    unsigned short* xc_bf    = w_out_bf + (size_t)DM_ * DI_;        // ML x DI
    unsigned short* xdbl_bf  = xc_bf   + (size_t)ML_ * DI_;         // ML x DX
    unsigned short* y_bf     = xdbl_bf + (size_t)ML_ * DX_;         // ML x DI

    const dim3 blk(256);

    // ---- pre-convert inputs to bf16 (one pass each) ----
    cvt_bf16_kernel<<<cdiv_((size_t)ML_ * DM_, 256), blk, 0, stream>>>(hs, hs_bf, (size_t)ML_ * DM_);
    cvt_bf16_kernel<<<cdiv_((size_t)2 * DI_ * DM_, 256), blk, 0, stream>>>(w_in, w_in_bf, (size_t)2 * DI_ * DM_);
    cvt_bf16_kernel<<<cdiv_((size_t)DX_ * DI_, 256), blk, 0, stream>>>(w_xp, w_xp_bf, (size_t)DX_ * DI_);
    cvt_bf16_kernel<<<cdiv_((size_t)DI_ * DR_, 256), blk, 0, stream>>>(w_dt, w_dt_bf, (size_t)DI_ * DR_);
    cvt_bf16_kernel<<<cdiv_((size_t)DM_ * DI_, 256), blk, 0, stream>>>(w_out, w_out_bf, (size_t)DM_ * DI_);

    // 1) xz = hs @ w_in^T              (N=8192, K=2048)
    launch_gemm<0>(hs_bf, DM_, w_in_bf, nullptr, xz, 2 * DI_, 2 * DI_, DM_, stream);
    // 2) depthwise conv + SiLU
    conv_silu_kernel<<<cdiv_((size_t)ML_ * DI_, 256), blk, 0, stream>>>(xz, w_conv, xc, xc_bf);
    // 3) x_dbl = xc @ w_xp^T           (N=160, K=4096)
    launch_gemm<0>(xc_bf, DI_, w_xp_bf, nullptr, xdbl, DX_, DX_, DI_, stream);
    cvt_bf16_kernel<<<cdiv_((size_t)ML_ * DX_, 256), blk, 0, stream>>>(xdbl, xdbl_bf, (size_t)ML_ * DX_);
    // 4) delta = softplus(dt_low @ w_dt^T + b)  (N=4096, K=128, lda=DX)
    launch_gemm<1>(xdbl_bf, DX_, w_dt_bf, b_dt, delta, DI_, DI_, DR_, stream);
    // 5) selective scan + D skip + SiLU(z) gate -> y (bf16)
    scan_kernel<<<dim3((B_ * DI_ + 255) / 256), blk, 0, stream>>>(
        delta, xdbl, xc, xz, A_log, Dp, y_bf);
    // 6) out = y @ w_out^T             (N=2048, K=4096)
    launch_gemm<0>(y_bf, DI_, w_out_bf, nullptr, out, DM_, DM_, DI_, stream);
}
